// Get_DefCorrelation_45664092291266
// MI455X (gfx1250) — matrix operand order repaired and checked by reading, not compiled, not run on hardware
//
#include <hip/hip_runtime.h>
#include <math.h>

// ---- problem constants (from reference) ----
#define CC   64
#define TT   16
#define HH   48
#define WWID 48
#define H2D  24
#define W2D  24
#define HWP  (HH*WWID)     // 2304 = P
#define SSD  (H2D*W2D)     // 576  = S
#define NPD  (TT*HWP)      // 36864
#define RCD  4

typedef __attribute__((ext_vector_type(16))) __bf16 v16bf;
typedef __attribute__((ext_vector_type(8)))  float  v8f;

__device__ __forceinline__ float sigm(float v) { return 1.0f / (1.0f + expf(-v)); }

// ---- WMMA fragment builders (ISA 7.12.2 layouts, bf16 16x16x32) ----
// A: 16(M) x 32(K).  lane<16 holds K {0..7,16..23}; lane>=16 holds K {8..15,24..31}; M = lane&15
__device__ __forceinline__ v16bf frag_a(const float* As, int lda, int k0, int lane) {
  v16bf a;
  int m  = lane & 15;
  int kb = (lane >> 4) << 3;            // 0 or 8
#pragma unroll
  for (int i = 0; i < 16; ++i) {
    int k = k0 + kb + i + ((i >= 8) ? 8 : 0);
    a[i] = (__bf16)As[m * lda + k];
  }
  return a;
}
// B: 32(K) x 16(N).  lane<16 holds K {0..7,16..23}; lane>=16 holds K {8..15,24..31}; N = lane&15
__device__ __forceinline__ v16bf frag_b(const float* Bs, int ldb, int k0, int lane) {
  v16bf b;
  int n  = lane & 15;
  int kb = (lane >> 4) << 3;            // 0 or 8
#pragma unroll
  for (int i = 0; i < 16; ++i) {
    int k = k0 + kb + i + ((i >= 8) ? 8 : 0);
    b[i] = (__bf16)Bs[k * ldb + n];
  }
  return b;
}

// ===================== 1) x2 = w_dc2 . x  (64x64 @ 64xNP) =====================
__global__ __launch_bounds__(32) void k_gemm_dc2(const float* __restrict__ x,
                                                 const float* __restrict__ w,   // [64][64]
                                                 float* __restrict__ x2) {
  __shared__ float As[16 * 64];
  __shared__ float Bs[64 * 16];
  int lane = threadIdx.x;
  int mo = blockIdx.y * 16;             // output-channel tile
  int n0 = blockIdx.x * 16;             // spatial tile
  for (int i = lane; i < 16 * 64; i += 32) As[i] = w[(mo + (i >> 6)) * CC + (i & 63)];
  for (int i = lane; i < 64 * 16; i += 32) {
    int c = i >> 4, j = i & 15;
    Bs[i] = x[(size_t)c * NPD + n0 + j];
  }
  __syncthreads();
  v8f acc = {};
  for (int k0 = 0; k0 < 64; k0 += 32) {
    v16bf a = frag_a(As, 64, k0, lane);
    v16bf b = frag_b(Bs, 16, k0, lane);
    acc = __builtin_amdgcn_wmma_f32_16x16x32_bf16(false, a, false, b, (short)0, acc, false, false);
  }
  int n = lane & 15, mb = (lane >> 4) * 8;
#pragma unroll
  for (int r = 0; r < 8; ++r) x2[(size_t)(mo + mb + r) * NPD + n0 + n] = acc[r];
}

// ===================== 2) offset convs (stride2, pad1), input = x + shifted x2 =====================
__global__ void k_offsets(const float* __restrict__ x, const float* __restrict__ x2,
                          const float* __restrict__ wl, const float* __restrict__ bl,
                          const float* __restrict__ wr, const float* __restrict__ br,
                          float* __restrict__ off) {
  int idx = blockIdx.x * blockDim.x + threadIdx.x;
  if (idx >= 2 * TT * 2 * SSD) return;
  int ox = idx % W2D; int tmp = idx / W2D;
  int oy = tmp % H2D; tmp /= H2D;
  int och = tmp & 1;  tmp >>= 1;
  int t = tmp % TT;   int side = tmp / TT;
  const float* wgt = side ? wr : wl;
  const float* bia = side ? br : bl;
  int ts = side ? (t > 0 ? t - 1 : 0) : (t < TT - 1 ? t + 1 : TT - 1);
  float acc = bia[och];
  for (int c = 0; c < CC; ++c) {
    const float* xp  = x  + ((size_t)c * TT + t ) * HWP;
    const float* x2p = x2 + ((size_t)c * TT + ts) * HWP;
#pragma unroll
    for (int ky = 0; ky < 3; ++ky) {
      int iy = oy * 2 - 1 + ky;
      if (iy < 0 || iy >= HH) continue;
#pragma unroll
      for (int kx = 0; kx < 3; ++kx) {
        int ix = ox * 2 - 1 + kx;
        if (ix < 0 || ix >= WWID) continue;
        acc += wgt[((och * CC + c) * 3 + ky) * 3 + kx] * (xp[iy * WWID + ix] + x2p[iy * WWID + ix]);
      }
    }
  }
  off[idx] = acc;
}

// ===================== 3) grid_sample(shifted x2, base*2 + off) -> part =====================
__global__ void k_sample(const float* __restrict__ x2, const float* __restrict__ off,
                         float* __restrict__ part) {
  int idx = blockIdx.x * blockDim.x + threadIdx.x;
  if (idx >= 2 * TT * CC * SSD) return;
  int sx = idx % W2D; int tmp = idx / W2D;
  int sy = tmp % H2D; tmp /= H2D;
  int c = tmp % CC;   tmp /= CC;
  int t = tmp % TT;   int side = tmp / TT;
  int ts = side ? (t > 0 ? t - 1 : 0) : (t < TT - 1 ? t + 1 : TT - 1);
  const float* img  = x2 + ((size_t)c * TT + ts) * HWP;
  const float* offp = off + (((size_t)side * TT + t) * 2) * SSD;
  float o0 = offp[sy * W2D + sx];
  float o1 = offp[SSD + sy * W2D + sx];
  // base = meshgrid * 2 ; faithful normalize_grid quirk (ch0 / (H2-1), ch1 / (W2-1))
  float v0 = 2.0f * (float)sx + o0;
  float v1 = 2.0f * (float)sy + o1;
  float gx = 2.0f * v0 / (float)(H2D - 1) - 1.0f;
  float gy = 2.0f * v1 / (float)(W2D - 1) - 1.0f;
  float fx = ((gx + 1.0f) * WWID - 1.0f) * 0.5f;
  float fy = ((gy + 1.0f) * HH   - 1.0f) * 0.5f;
  int x0 = (int)floorf(fx), y0 = (int)floorf(fy);
  float dx = fx - (float)x0, dy = fy - (float)y0;
  float res = 0.0f;
  int   xs[4] = {x0, x0 + 1, x0, x0 + 1};
  int   ys[4] = {y0, y0, y0 + 1, y0 + 1};
  float wc[4] = {(1.0f - dx) * (1.0f - dy), dx * (1.0f - dy), (1.0f - dx) * dy, dx * dy};
#pragma unroll
  for (int k = 0; k < 4; ++k) {
    int xi = xs[k], yi = ys[k];
    if (xi >= 0 && xi < WWID && yi >= 0 && yi < HH) res += wc[k] * img[yi * WWID + xi];
  }
  part[idx] = res;
}

// ===================== 4) aff = X_t^T . Part_t  (P x S, K=64) =====================
__global__ __launch_bounds__(32) void k_aff(const float* __restrict__ x,
                                            const float* __restrict__ partp,   // [C][S] for (side,t)
                                            float* __restrict__ aff, int t) {
  __shared__ float As[16 * 64];   // A[p][c]
  __shared__ float Bs[64 * 16];   // B[c][s]
  int lane = threadIdx.x;
  int p0 = blockIdx.x * 16;
  int s0 = blockIdx.y * 16;
  for (int i = lane; i < 16 * 64; i += 32) {
    int pi = i >> 6, c = i & 63;
    As[i] = x[((size_t)c * TT + t) * HWP + p0 + pi];
  }
  for (int i = lane; i < 64 * 16; i += 32) {
    int c = i >> 4, j = i & 15;
    Bs[i] = partp[(size_t)c * SSD + s0 + j];
  }
  __syncthreads();
  v8f acc = {};
  for (int k0 = 0; k0 < 64; k0 += 32) {
    v16bf a = frag_a(As, 64, k0, lane);
    v16bf b = frag_b(Bs, 16, k0, lane);
    acc = __builtin_amdgcn_wmma_f32_16x16x32_bf16(false, a, false, b, (short)0, acc, false, false);
  }
  int n = lane & 15, mb = (lane >> 4) * 8;
#pragma unroll
  for (int r = 0; r < 8; ++r) aff[(size_t)(p0 + mb + r) * SSD + s0 + n] = acc[r];
}

// ===================== 5) channel stats over S (mean / max / unbiased var) =====================
__global__ void k_stats(const float* __restrict__ aff, float* __restrict__ avg,
                        float* __restrict__ mx, float* __restrict__ var) {
  int p = blockIdx.x * blockDim.x + threadIdx.x;
  if (p >= HWP) return;
  const float* row = aff + (size_t)p * SSD;
  float sum = 0.0f, m = -3.4e38f, sq = 0.0f;
  for (int s = 0; s < SSD; ++s) { float v = row[s]; sum += v; sq += v * v; m = fmaxf(m, v); }
  float mean = sum / (float)SSD;
  avg[p] = mean; mx[p] = m;
  var[p] = (sq - (float)SSD * mean * mean) / (float)(SSD - 1);
}

// ===================== 6) attention 3x3 convs -> y = sigmoid(conv([avg,mx]) + conv(var)) =====================
__global__ void k_attn(const float* __restrict__ avg, const float* __restrict__ mx,
                       const float* __restrict__ var, const float* __restrict__ w1,
                       const float* __restrict__ w2, float* __restrict__ y) {
  int p = blockIdx.x * blockDim.x + threadIdx.x;
  if (p >= HWP) return;
  int h = p / WWID, w = p % WWID;
  float acc = 0.0f;
#pragma unroll
  for (int ky = 0; ky < 3; ++ky) {
    int ih = h + ky - 1;
    if (ih < 0 || ih >= HH) continue;
#pragma unroll
    for (int kx = 0; kx < 3; ++kx) {
      int iw = w + kx - 1;
      if (iw < 0 || iw >= WWID) continue;
      int q = ih * WWID + iw;
      acc += w1[ky * 3 + kx] * avg[q] + w1[9 + ky * 3 + kx] * mx[q] + w2[ky * 3 + kx] * var[q];
    }
  }
  y[p] = sigm(acc);
}

// ===================== 7) features += w2v[side] * Part . (sigmoid(aff*y)-0.5)^T  (C x P, K=S) =====================
__global__ __launch_bounds__(32) void k_out(const float* __restrict__ partp,   // [C][S]
                                            const float* __restrict__ aff,     // [P][S]
                                            const float* __restrict__ y,       // [P]
                                            float* __restrict__ feat,          // [C][NP]
                                            const float* __restrict__ w2v, int side, int t) {
  __shared__ float As[16 * 32];   // A[c][k]
  __shared__ float Bs[32 * 16];   // B[k][p]
  int lane = threadIdx.x;
  int p0 = blockIdx.x * 16;
  int c0 = blockIdx.y * 16;
  v8f acc = {};
  for (int ks = 0; ks < SSD; ks += 32) {
    for (int i = lane; i < 16 * 32; i += 32) {
      int ci = i >> 5, kk = i & 31;
      As[i] = partp[(size_t)(c0 + ci) * SSD + ks + kk];
    }
    for (int i = lane; i < 32 * 16; i += 32) {
      int kk = i >> 4, j = i & 15;
      float a = aff[(size_t)(p0 + j) * SSD + ks + kk] * y[p0 + j];
      Bs[i] = sigm(a) - 0.5f;
    }
    __syncthreads();
    v16bf a = frag_a(As, 32, 0, lane);
    v16bf b = frag_b(Bs, 16, 0, lane);
    acc = __builtin_amdgcn_wmma_f32_16x16x32_bf16(false, a, false, b, (short)0, acc, false, false);
    __syncthreads();
  }
  float ws = w2v[side];
  int n = lane & 15, mb = (lane >> 4) * 8;
#pragma unroll
  for (int r = 0; r < 8; ++r) {
    size_t o = (size_t)(c0 + mb + r) * NPD + (size_t)t * HWP + p0 + n;
    feat[o] += ws * acc[r];
  }
}

// ===================== 8) xd = w_down . x =====================
__global__ void k_down(const float* __restrict__ x, const float* __restrict__ wd,
                       float* __restrict__ xd) {
  int idx = blockIdx.x * blockDim.x + threadIdx.x;
  if (idx >= RCD * NPD) return;
  int rc = idx / NPD, p = idx % NPD;
  float acc = 0.0f;
  for (int c = 0; c < CC; ++c) acc += wd[rc * CC + c] * x[(size_t)c * NPD + p];
  xd[idx] = acc;
}

// ===================== 9) three dilated depthwise 9x3x3 convs, weighted sum =====================
__global__ void k_agg(const float* __restrict__ xd,
                      const float* __restrict__ w1, const float* __restrict__ b1,
                      const float* __restrict__ w2, const float* __restrict__ b2,
                      const float* __restrict__ w3, const float* __restrict__ b3,
                      const float* __restrict__ wts, float* __restrict__ agg) {
  int idx = blockIdx.x * blockDim.x + threadIdx.x;
  if (idx >= RCD * NPD) return;
  int w = idx % WWID; int tmp = idx / WWID;
  int h = tmp % HH; tmp /= HH;
  int t = tmp % TT; int rc = tmp / TT;
  const float* wa[3] = {w1, w2, w3};
  const float* ba[3] = {b1, b2, b3};
  float out = 0.0f;
  for (int i = 0; i < 3; ++i) {
    int dil = i + 1;
    float acc = ba[i][rc];
    for (int kt = 0; kt < 9; ++kt) {
      int tt = t + kt - 4;
      if (tt < 0 || tt >= TT) continue;
#pragma unroll
      for (int ky = 0; ky < 3; ++ky) {
        int hh = h + dil * (ky - 1);
        if (hh < 0 || hh >= HH) continue;
#pragma unroll
        for (int kx = 0; kx < 3; ++kx) {
          int ww = w + dil * (kx - 1);
          if (ww < 0 || ww >= WWID) continue;
          acc += wa[i][((rc * 9 + kt) * 3 + ky) * 3 + kx] *
                 xd[((size_t)rc * TT + tt) * HWP + hh * WWID + ww];
        }
      }
    }
    out += wts[i] * acc;
  }
  agg[idx] = out;
}

// ===================== 10) out = features * (sigmoid(w_back . agg) - 0.5) =====================
__global__ void k_final(const float* __restrict__ feat, const float* __restrict__ agg,
                        const float* __restrict__ wb, float* __restrict__ out) {
  int idx = blockIdx.x * blockDim.x + threadIdx.x;
  if (idx >= CC * NPD) return;
  int c = idx / NPD, tp = idx % NPD;
  float a = 0.0f;
#pragma unroll
  for (int rc = 0; rc < RCD; ++rc) a += wb[c * RCD + rc] * agg[(size_t)rc * NPD + tp];
  out[idx] = feat[idx] * (sigm(a) - 0.5f);
}

// ===================================================================================
extern "C" void kernel_launch(void* const* d_in, const int* in_sizes, int n_in,
                              void* d_out, int out_size, void* d_ws, size_t ws_size,
                              hipStream_t stream) {
  const float* x       = (const float*)d_in[0];
  const float* w_dc2   = (const float*)d_in[1];
  const float* w_ofs_l = (const float*)d_in[2];
  const float* b_ofs_l = (const float*)d_in[3];
  const float* w_ofs_r = (const float*)d_in[4];
  const float* b_ofs_r = (const float*)d_in[5];
  const float* ca_w1   = (const float*)d_in[6];
  const float* ca_w2   = (const float*)d_in[7];
  const float* w_down  = (const float*)d_in[8];
  const float* sa1_w   = (const float*)d_in[9];
  const float* sa1_b   = (const float*)d_in[10];
  const float* sa2_w   = (const float*)d_in[11];
  const float* sa2_b   = (const float*)d_in[12];
  const float* sa3_w   = (const float*)d_in[13];
  const float* sa3_b   = (const float*)d_in[14];
  const float* weights  = (const float*)d_in[15];
  const float* weights2 = (const float*)d_in[16];
  const float* w_back   = (const float*)d_in[17];
  float* out = (float*)d_out;

  // workspace carve-up (floats)
  float* ws   = (float*)d_ws;
  float* x2   = ws;                                  // 64*NP          = 2359296
  float* off  = x2 + (size_t)CC * NPD;               // 2*16*2*576    = 36864
  float* part = off + (size_t)2 * TT * 2 * SSD;      // 2*16*64*576   = 1179648
  float* aff  = part + (size_t)2 * TT * CC * SSD;    // 2304*576      = 1327104
  float* avg  = aff + (size_t)HWP * SSD;             // 2304
  float* mx   = avg + HWP;
  float* var  = mx + HWP;
  float* ymap = var + HWP;
  float* feat = ymap + HWP;                          // 64*NP
  float* xd   = feat + (size_t)CC * NPD;             // 4*NP
  float* agg  = xd + (size_t)RCD * NPD;              // 4*NP

  hipMemsetAsync(feat, 0, sizeof(float) * (size_t)CC * NPD, stream);

  // 1) x2 = w_dc2 . x   (WMMA)
  k_gemm_dc2<<<dim3(NPD / 16, CC / 16), 32, 0, stream>>>(x, w_dc2, x2);

  // 2) offsets, 3) grid-sample
  {
    int n = 2 * TT * 2 * SSD;
    k_offsets<<<(n + 255) / 256, 256, 0, stream>>>(x, x2, w_ofs_l, b_ofs_l, w_ofs_r, b_ofs_r, off);
  }
  {
    int n = 2 * TT * CC * SSD;
    k_sample<<<(n + 255) / 256, 256, 0, stream>>>(x2, off, part);
  }

  // 8)-9) spatio-temporal aggregation path (independent of correlation path)
  {
    int n = RCD * NPD;
    k_down<<<(n + 255) / 256, 256, 0, stream>>>(x, w_down, xd);
    k_agg<<<(n + 255) / 256, 256, 0, stream>>>(xd, sa1_w, sa1_b, sa2_w, sa2_b, sa3_w, sa3_b,
                                               weights, agg);
  }

  // 4)-7) correlation branches (WMMA-heavy)
  for (int side = 0; side < 2; ++side) {
    for (int t = 0; t < TT; ++t) {
      const float* partp = part + (((size_t)side * TT + t) * CC) * SSD;
      k_aff<<<dim3(HWP / 16, SSD / 16), 32, 0, stream>>>(x, partp, aff, t);
      k_stats<<<(HWP + 255) / 256, 256, 0, stream>>>(aff, avg, mx, var);
      k_attn<<<(HWP + 255) / 256, 256, 0, stream>>>(avg, mx, var, ca_w1, ca_w2, ymap);
      k_out<<<dim3(HWP / 16, CC / 16), 32, 0, stream>>>(partp, aff, ymap, feat, weights2, side, t);
    }
  }

  // 10) final gating
  {
    int n = CC * NPD;
    k_final<<<(n + 255) / 256, 256, 0, stream>>>(feat, agg, w_back, out);
  }
}
